// MHA_6751688589728
// MI455X (gfx1250) — matrix operand order repaired and checked by reading, not compiled, not run on hardware
//
#include <hip/hip_runtime.h>

// ---------------------------------------------------------------------------
// MHA for MI455X (gfx1250): bf16 WMMA (v_wmma_f32_16x16x32_bf16) for all three
// GEMM stages, fp32 flash-softmax in VALU. Wave32, 8 waves / 256-thread block.
// Round 2: 128x128 GEMM block tiles (32x64 per wave -> 8 WMMA per K-step per
// wave), fragment preloading so WMMAs issue back-to-back, 128-row attention
// tiles, vectorized V^T staging.
// ---------------------------------------------------------------------------

typedef __bf16 v16bf __attribute__((ext_vector_type(16)));
typedef float  v8f   __attribute__((ext_vector_type(8)));

union FragAB { v16bf v; uint4 q[2]; };

__device__ __forceinline__ unsigned short f2bf(float f) {
    unsigned int u = __builtin_bit_cast(unsigned int, f);
    u += 0x7FFFu + ((u >> 16) & 1u);   // round-to-nearest-even
    return (unsigned short)(u >> 16);
}

static constexpr int Bsz = 8;
static constexpr int Msz = 1024;
static constexpr int Dsz = 1024;
static constexpr int Hn  = 16;
static constexpr int Kd  = 64;
static constexpr size_t QKV_ELEMS = (size_t)Bsz * Hn * Msz * Kd;   // 8,388,608

// A-fragment (16x32 bf16) from an LDS row of 32 bf16 (4 uint4 per row)
__device__ __forceinline__ FragAB loadA32(const unsigned short (*T)[32], int row, int hi) {
    FragAB a;
    const uint4* p = (const uint4*)&T[row][0];
    a.q[0] = p[hi];
    a.q[1] = p[2 + hi];
    return a;
}
// B-fragment (32x16 bf16): lanes 0-15 K=0..7,16..23; lanes 16-31 K=8..15,24..31
__device__ __forceinline__ FragAB loadB32(const unsigned short (*T)[32], int row, int hi) {
    FragAB b;
    const uint4* p = (const uint4*)&T[row][0];
    b.q[0] = p[hi];
    b.q[1] = p[2 + hi];
    return b;
}
// Same loaders for 64-deep LDS rows (8 uint4 per row), selecting 32-wide K step
__device__ __forceinline__ FragAB loadA64(const unsigned short (*T)[64], int row, int ks, int hi) {
    FragAB a;
    const uint4* p = (const uint4*)&T[row][0];
    a.q[0] = p[4 * ks + hi];
    a.q[1] = p[4 * ks + 2 + hi];
    return a;
}
__device__ __forceinline__ FragAB loadB64(const unsigned short (*T)[64], int row, int ks, int hi) {
    FragAB b;
    const uint4* p = (const uint4*)&T[row][0];
    b.q[0] = p[4 * ks + hi];
    b.q[1] = p[4 * ks + 2 + hi];
    return b;
}

#define WMMA_BF16(A, B, C) \
    __builtin_amdgcn_wmma_f32_16x16x32_bf16(false, (A).v, false, (B).v, (short)0, (C), false, false)

// ---------------------------------------------------------------------------
// Kernel 1: t = x @ P_i  (8192x1024 @ 1024x3072) -> Q/K/V [b][h][m][64] bf16.
// 256 threads, 128x128 tile; wave (wm,wn) owns a 32x64 strip: 2x4 WMMA tiles.
// ---------------------------------------------------------------------------
__global__ __launch_bounds__(256) void qkv_proj_kernel(
    const float* __restrict__ x, const float* __restrict__ Pi,
    unsigned short* __restrict__ Qw, unsigned short* __restrict__ Kw,
    unsigned short* __restrict__ Vw)
{
    __shared__ alignas(16) unsigned short At[128][32];   // [m][k]
    __shared__ alignas(16) unsigned short Bt[128][32];   // [n][k] (P_i^T tile)

    const int m0   = blockIdx.x * 128;
    const int n0   = blockIdx.y * 128;
    const int tid  = threadIdx.x;
    const int wv   = tid >> 5;
    const int wm   = wv & 3;          // M group: rows wm*32..+31
    const int wn   = wv >> 2;         // N group: cols wn*64..+63
    const int lane = tid & 31;
    const int lcol = lane & 15;
    const int hi   = lane >> 4;

    v8f acc[2][4];
    #pragma unroll
    for (int mi = 0; mi < 2; ++mi)
        #pragma unroll
        for (int nt = 0; nt < 4; ++nt)
            #pragma unroll
            for (int j = 0; j < 8; ++j) acc[mi][nt][j] = 0.0f;

    for (int K0 = 0; K0 < Dsz; K0 += 32) {
        __syncthreads();
        __builtin_prefetch(&x[(size_t)(m0 + (tid >> 1)) * Dsz + K0 + 32], 0, 0);
        // stage A (coalesced 32-wide rows), fp32->bf16: 4096 elems / 256 thr
        for (int e = tid; e < 128 * 32; e += 256) {
            int i = e >> 5, k = e & 31;
            At[i][k] = f2bf(x[(size_t)(m0 + i) * Dsz + K0 + k]);
        }
        // stage B transposed (coalesced 128-wide over n)
        for (int e = tid; e < 32 * 128; e += 256) {
            int k = e >> 7, n = e & 127;
            Bt[n][k] = f2bf(Pi[(size_t)(K0 + k) * (3 * Dsz) + n0 + n]);
        }
        __syncthreads();

        // preload all fragments, then issue the 8-WMMA chain back-to-back
        FragAB a0 = loadA32(At, wm * 32 + lcol, hi);
        FragAB a1 = loadA32(At, wm * 32 + 16 + lcol, hi);
        FragAB b0 = loadB32(Bt, wn * 64 + 0 * 16 + lcol, hi);
        FragAB b1 = loadB32(Bt, wn * 64 + 1 * 16 + lcol, hi);
        FragAB b2 = loadB32(Bt, wn * 64 + 2 * 16 + lcol, hi);
        FragAB b3 = loadB32(Bt, wn * 64 + 3 * 16 + lcol, hi);
        acc[0][0] = WMMA_BF16(a0, b0, acc[0][0]);
        acc[0][1] = WMMA_BF16(a0, b1, acc[0][1]);
        acc[0][2] = WMMA_BF16(a0, b2, acc[0][2]);
        acc[0][3] = WMMA_BF16(a0, b3, acc[0][3]);
        acc[1][0] = WMMA_BF16(a1, b0, acc[1][0]);
        acc[1][1] = WMMA_BF16(a1, b1, acc[1][1]);
        acc[1][2] = WMMA_BF16(a1, b2, acc[1][2]);
        acc[1][3] = WMMA_BF16(a1, b3, acc[1][3]);
    }

    // scatter C into Q/K/V: output col c -> h=c/192, sel=(c%192)/64, kk=c%64
    #pragma unroll
    for (int mi = 0; mi < 2; ++mi) {
        const int mrowbase = m0 + wm * 32 + mi * 16;
        #pragma unroll
        for (int nt = 0; nt < 4; ++nt) {
            int col = n0 + wn * 64 + nt * 16 + lcol;
            int h   = col / 192;
            int rem = col % 192;
            int sel = rem / 64;
            int kk  = rem % 64;
            unsigned short* dst = (sel == 0) ? Qw : (sel == 1) ? Kw : Vw;
            #pragma unroll
            for (int r = 0; r < 8; ++r) {
                int mrow = mrowbase + r + 8 * hi;
                int b    = mrow >> 10;
                int mloc = mrow & 1023;
                dst[((size_t)(b * Hn + h) * Msz + mloc) * Kd + kk] = f2bf(acc[mi][nt][r]);
            }
        }
    }
}

// ---------------------------------------------------------------------------
// Kernel 2: flash attention per (b,h, 128-row tile). Causal, online softmax.
// 256 threads / 8 waves; each wave owns a 16x64 strip. K/V staging amortized
// over 128 query rows.
// ---------------------------------------------------------------------------
__global__ __launch_bounds__(256) void attn_kernel(
    const unsigned short* __restrict__ Qw, const unsigned short* __restrict__ Kw,
    const unsigned short* __restrict__ Vw, unsigned short* __restrict__ Ow)
{
    __shared__ alignas(16) unsigned short Qt[128][64];  // [m][k]
    __shared__ alignas(16) unsigned short Kt[64][64];   // [n][k]  (B for QK^T)
    __shared__ alignas(16) unsigned short Vt[64][64];   // [kk][n] (V^T, B for PV)
    __shared__ alignas(16) unsigned short Pt[128][64];  // [m][n]  (probs, bf16)

    const int tid  = threadIdx.x;
    const int wv   = tid >> 5;
    const int lane = tid & 31;
    const int lcol = lane & 15;
    const int hi   = lane >> 4;

    const int mtile = blockIdx.x & 7;       // 8 tiles of 128 rows
    const int bh    = blockIdx.x >> 3;      // b*16+h
    const int m0    = mtile * 128;
    const size_t base = (size_t)bh * Msz * Kd;

    // stage the Q tile once (bf16; 128-bit LDS stores): 1024 chunks / 256 thr
    for (int c = tid; c < 128 * 8; c += 256) {
        int row = c >> 3, q = c & 7;
        ((uint4*)&Qt[row][0])[q] =
            ((const uint4*)(Qw + base + (size_t)(m0 + row) * Kd))[q];
    }

    float mrun[8], lrun[8];
    v8f O[4];
    #pragma unroll
    for (int r = 0; r < 8; ++r) { mrun[r] = -1e30f; lrun[r] = 0.0f; }
    #pragma unroll
    for (int t = 0; t < 4; ++t)
        #pragma unroll
        for (int j = 0; j < 8; ++j) O[t][j] = 0.0f;

    const int mrow_base = m0 + wv * 16;
    const int qrow = wv * 16 + lcol;

    for (int n0 = 0; n0 <= m0 + 64; n0 += 64) {
        __syncthreads();
        __builtin_prefetch(Kw + base + (size_t)(n0 + 64 + (tid >> 2)) * Kd, 0, 0);
        // K tile: 512 uint4 chunks / 256 thr
        for (int c = tid; c < 512; c += 256) {
            int row = c >> 3, q = c & 7;
            ((uint4*)&Kt[row][0])[q] =
                ((const uint4*)(Kw + base + (size_t)(n0 + row) * Kd))[q];
        }
        // V^T tile: vectorized row loads, scalar transposed LDS stores
        for (int c = tid; c < 512; c += 256) {
            int n = c >> 3, q = c & 7;
            uint4 vv = ((const uint4*)(Vw + base + (size_t)(n0 + n) * Kd))[q];
            const unsigned short* pv = (const unsigned short*)&vv;
            #pragma unroll
            for (int j = 0; j < 8; ++j) Vt[q * 8 + j][n] = pv[j];
        }
        __syncthreads();

        // ---- S = Q K^T (16x64 strip, K-depth 64 = 2 WMMA steps)
        v8f S[4];
        #pragma unroll
        for (int t = 0; t < 4; ++t)
            #pragma unroll
            for (int j = 0; j < 8; ++j) S[t][j] = 0.0f;
        #pragma unroll
        for (int ks = 0; ks < 2; ++ks) {
            FragAB a  = loadA64(Qt, qrow, ks, hi);
            FragAB b0 = loadB64(Kt, 0 * 16 + lcol, ks, hi);
            FragAB b1 = loadB64(Kt, 1 * 16 + lcol, ks, hi);
            FragAB b2 = loadB64(Kt, 2 * 16 + lcol, ks, hi);
            FragAB b3 = loadB64(Kt, 3 * 16 + lcol, ks, hi);
            S[0] = WMMA_BF16(a, b0, S[0]);
            S[1] = WMMA_BF16(a, b1, S[1]);
            S[2] = WMMA_BF16(a, b2, S[2]);
            S[3] = WMMA_BF16(a, b3, S[3]);
        }

        // ---- scale + causal mask
        #pragma unroll
        for (int nt = 0; nt < 4; ++nt) {
            int ncol = n0 + nt * 16 + lcol;
            #pragma unroll
            for (int r = 0; r < 8; ++r) {
                int mrow = mrow_base + r + 8 * hi;
                float s = S[nt][r] * 0.125f;                 // 1/sqrt(64)
                S[nt][r] = (ncol <= mrow) ? s : -1e30f;
            }
        }

        // ---- online softmax (row = r + 8*hi; shfl reductions stay row-aligned)
        #pragma unroll
        for (int r = 0; r < 8; ++r) {
            float v = fmaxf(fmaxf(S[0][r], S[1][r]), fmaxf(S[2][r], S[3][r]));
            v = fmaxf(v, __shfl_xor(v, 1));
            v = fmaxf(v, __shfl_xor(v, 2));
            v = fmaxf(v, __shfl_xor(v, 4));
            v = fmaxf(v, __shfl_xor(v, 8));
            float mnew = fmaxf(mrun[r], v);
            float corr = __expf(mrun[r] - mnew);
            float rsum = 0.0f;
            #pragma unroll
            for (int nt = 0; nt < 4; ++nt) {
                float p = __expf(S[nt][r] - mnew);
                S[nt][r] = p;
                rsum += p;
            }
            rsum += __shfl_xor(rsum, 1);
            rsum += __shfl_xor(rsum, 2);
            rsum += __shfl_xor(rsum, 4);
            rsum += __shfl_xor(rsum, 8);
            lrun[r] = lrun[r] * corr + rsum;
            mrun[r] = mnew;
            #pragma unroll
            for (int nt = 0; nt < 4; ++nt) O[nt][r] *= corr;
        }

        // ---- write P strip (C-layout -> LDS) so it can be re-read as A-frags
        #pragma unroll
        for (int nt = 0; nt < 4; ++nt)
            #pragma unroll
            for (int r = 0; r < 8; ++r)
                Pt[wv * 16 + r + 8 * hi][nt * 16 + lcol] = f2bf(S[nt][r]);
        __syncthreads();

        // ---- O += P V (contraction over n, 2 WMMA steps)
        #pragma unroll
        for (int ks = 0; ks < 2; ++ks) {
            FragAB a  = loadA64(Pt, qrow, ks, hi);
            FragAB b0 = loadB64(Vt, 0 * 16 + lcol, ks, hi);
            FragAB b1 = loadB64(Vt, 1 * 16 + lcol, ks, hi);
            FragAB b2 = loadB64(Vt, 2 * 16 + lcol, ks, hi);
            FragAB b3 = loadB64(Vt, 3 * 16 + lcol, ks, hi);
            O[0] = WMMA_BF16(a, b0, O[0]);
            O[1] = WMMA_BF16(a, b1, O[1]);
            O[2] = WMMA_BF16(a, b2, O[2]);
            O[3] = WMMA_BF16(a, b3, O[3]);
        }
    }

    // ---- normalize and store o in [b][m][h*64+kk] layout (row = b*1024+m)
    const int b_idx = bh >> 4, h = bh & 15;
    #pragma unroll
    for (int nt = 0; nt < 4; ++nt) {
        int kk = nt * 16 + lcol;
        #pragma unroll
        for (int r = 0; r < 8; ++r) {
            int   mloc = m0 + wv * 16 + r + 8 * hi;
            float inv  = 1.0f / lrun[r];
            Ow[((size_t)(b_idx * Msz + mloc)) * Dsz + h * Kd + kk] =
                f2bf(O[nt][r] * inv);
        }
    }
}

// ---------------------------------------------------------------------------
// Kernel 3: y = o @ P_o  (8192x1024 @ 1024x1024), fp32 output.
// Same 128x128 / 32x64-per-wave tiling as kernel 1.
// ---------------------------------------------------------------------------
__global__ __launch_bounds__(256) void out_proj_kernel(
    const unsigned short* __restrict__ Ow, const float* __restrict__ Po,
    float* __restrict__ y)
{
    __shared__ alignas(16) unsigned short At[128][32];
    __shared__ alignas(16) unsigned short Bt[128][32];

    const int m0   = blockIdx.x * 128;
    const int n0   = blockIdx.y * 128;
    const int tid  = threadIdx.x;
    const int wv   = tid >> 5;
    const int wm   = wv & 3;
    const int wn   = wv >> 2;
    const int lane = tid & 31;
    const int lcol = lane & 15;
    const int hi   = lane >> 4;

    v8f acc[2][4];
    #pragma unroll
    for (int mi = 0; mi < 2; ++mi)
        #pragma unroll
        for (int nt = 0; nt < 4; ++nt)
            #pragma unroll
            for (int j = 0; j < 8; ++j) acc[mi][nt][j] = 0.0f;

    for (int K0 = 0; K0 < Dsz; K0 += 32) {
        __syncthreads();
        // stage A (already bf16): 512 uint4 chunks / 256 thr
        for (int c = tid; c < 512; c += 256) {
            int row = c >> 2, q = c & 3;
            ((uint4*)&At[row][0])[q] =
                ((const uint4*)(Ow + (size_t)(m0 + row) * Dsz + K0))[q];
        }
        // stage B transposed, fp32->bf16 (coalesced 128-wide over n)
        for (int e = tid; e < 32 * 128; e += 256) {
            int k = e >> 7, n = e & 127;
            Bt[n][k] = f2bf(Po[(size_t)(K0 + k) * Dsz + n0 + n]);
        }
        __syncthreads();

        FragAB a0 = loadA32(At, wm * 32 + lcol, hi);
        FragAB a1 = loadA32(At, wm * 32 + 16 + lcol, hi);
        FragAB b0 = loadB32(Bt, wn * 64 + 0 * 16 + lcol, hi);
        FragAB b1 = loadB32(Bt, wn * 64 + 1 * 16 + lcol, hi);
        FragAB b2 = loadB32(Bt, wn * 64 + 2 * 16 + lcol, hi);
        FragAB b3 = loadB32(Bt, wn * 64 + 3 * 16 + lcol, hi);
        acc[0][0] = WMMA_BF16(a0, b0, acc[0][0]);
        acc[0][1] = WMMA_BF16(a0, b1, acc[0][1]);
        acc[0][2] = WMMA_BF16(a0, b2, acc[0][2]);
        acc[0][3] = WMMA_BF16(a0, b3, acc[0][3]);
        acc[1][0] = WMMA_BF16(a1, b0, acc[1][0]);
        acc[1][1] = WMMA_BF16(a1, b1, acc[1][1]);
        acc[1][2] = WMMA_BF16(a1, b2, acc[1][2]);
        acc[1][3] = WMMA_BF16(a1, b3, acc[1][3]);
    }

    #pragma unroll
    for (int mi = 0; mi < 2; ++mi) {
        #pragma unroll
        for (int nt = 0; nt < 4; ++nt) {
            int col = n0 + wn * 64 + nt * 16 + lcol;
            #pragma unroll
            for (int r = 0; r < 8; ++r) {
                int mrow = m0 + wm * 32 + mi * 16 + r + 8 * hi;
                y[(size_t)mrow * Dsz + col] = acc[mi][nt][r];
            }
        }
    }
}

// ---------------------------------------------------------------------------
extern "C" void kernel_launch(void* const* d_in, const int* in_sizes, int n_in,
                              void* d_out, int out_size, void* d_ws, size_t ws_size,
                              hipStream_t stream) {
    (void)in_sizes; (void)n_in; (void)out_size; (void)ws_size;
    const float* x  = (const float*)d_in[0];
    const float* Pi = (const float*)d_in[1];
    const float* Po = (const float*)d_in[2];
    float* y = (float*)d_out;

    // workspace: Q,K,V [b][h][m][64] bf16 (16MB each) + o [b*m][1024] bf16 (16MB)
    unsigned short* Qw = (unsigned short*)d_ws;
    unsigned short* Kw = Qw + QKV_ELEMS;
    unsigned short* Vw = Kw + QKV_ELEMS;
    unsigned short* Ow = Vw + QKV_ELEMS;

    qkv_proj_kernel<<<dim3(64, 24), 256, 0, stream>>>(x, Pi, Qw, Kw, Vw);
    attn_kernel<<<dim3(1024), 256, 0, stream>>>(Qw, Kw, Vw, Ow);
    out_proj_kernel<<<dim3(64, 8), 256, 0, stream>>>(Ow, Po, y);
}
